// GAT_46402826666070
// MI455X (gfx1250) — compile-verified, matrix operand order checked by reading, not run on hardware
//
#include <hip/hip_runtime.h>
#include <hip/hip_bf16.h>

#define HEADS 4
#define OUTC 64
#define FEAT 256           // IN == HEADS*OUTC == 256
#define NEG_SLOPE 0.2f

typedef __attribute__((ext_vector_type(16))) __bf16    v16bf;
typedef __attribute__((ext_vector_type(8)))  float     v8f;
typedef __attribute__((ext_vector_type(4)))  unsigned  v4u;

union AFrag { v4u u[2]; v16bf v; };
union CFrag { float f[8]; v8f v; };

// ---------- helpers ----------
__device__ inline unsigned short f2bf(float f) {
    unsigned u = __float_as_uint(f);
    unsigned r = u + 0x7FFFu + ((u >> 16) & 1u);   // round-to-nearest-even
    return (unsigned short)(r >> 16);
}
// monotonic float<->uint key for atomic max over signed floats
__device__ inline unsigned ordKey(float f) {
    unsigned u = __float_as_uint(f);
    return (u & 0x80000000u) ? ~u : (u | 0x80000000u);
}
__device__ inline float ordToFloat(unsigned k) {
    unsigned u = (k & 0x80000000u) ? (k & 0x7FFFFFFFu) : ~k;
    return __uint_as_float(u);
}
__device__ inline float edgeLogit(const float* __restrict__ a_src,
                                  const float* __restrict__ a_dst,
                                  int s, int d, int hh) {
    float e = a_src[s * HEADS + hh] + a_dst[d * HEADS + hh];
    return e >= 0.f ? e : NEG_SLOPE * e;
}

// ---------- init ----------
__global__ void zero_u32_kernel(unsigned* __restrict__ p, long long n) {
    long long i = (long long)blockIdx.x * blockDim.x + threadIdx.x;
    long long stride = (long long)gridDim.x * blockDim.x;
    for (; i < n; i += stride) p[i] = 0u;
}

// ---------- bf16 conversion ----------
__global__ void cvt_bf16_kernel(const float* __restrict__ src,
                                unsigned short* __restrict__ dst, long long n) {
    long long i = (long long)blockIdx.x * blockDim.x + threadIdx.x;
    long long stride = (long long)gridDim.x * blockDim.x;
    for (; i < n; i += stride) dst[i] = f2bf(src[i]);
}
// W[in,out] (row-major) -> Wt[out,in] bf16
__global__ void cvt_wT_kernel(const float* __restrict__ W,
                              unsigned short* __restrict__ wt) {
    int idx = blockIdx.x * 256 + threadIdx.x;     // FEAT*FEAT threads
    int o = idx >> 8, i = idx & 255;
    wt[o * FEAT + i] = f2bf(W[i * FEAT + o]);
}

// ---------- GEMM: h[N,256] = x[N,256] @ W[256,256], bf16 WMMA, fp32 acc ----------
__global__ void __launch_bounds__(256)
gemm_wmma_bf16_kernel(const unsigned short* __restrict__ xb,
                      const unsigned short* __restrict__ wt,
                      float* __restrict__ h) {
    const int wave = threadIdx.x >> 5;
    const int lane = threadIdx.x & 31;
    const int mt = blockIdx.x >> 1;                       // M tile
    const int nt = ((blockIdx.x & 1) << 3) + wave;        // N tile (16 per row, 8 waves/block)
    const int m0 = mt << 4, n0 = nt << 4;
    const int r  = lane & 15;       // M row (A) / N col (B,C)
    const int hi = lane >> 4;       // lane-half selects K sub-range / C row-half
    const int kb = hi << 3;

    const unsigned short* xrow = xb + (long long)(m0 + r) * FEAT;
    const unsigned short* wrow = wt + (long long)(n0 + r) * FEAT;

    CFrag acc;
#pragma unroll
    for (int i = 0; i < 8; ++i) acc.f[i] = 0.f;

#pragma unroll
    for (int k0 = 0; k0 < FEAT; k0 += 32) {
        if (k0 + 64 < FEAT) __builtin_prefetch(xrow + k0 + 64, 0, 1);
        AFrag a, b;
        a.u[0] = *(const v4u*)(xrow + k0 + kb);        // K = kb..kb+7
        a.u[1] = *(const v4u*)(xrow + k0 + kb + 16);   // K = kb+16..kb+23
        b.u[0] = *(const v4u*)(wrow + k0 + kb);
        b.u[1] = *(const v4u*)(wrow + k0 + kb + 16);
        acc.v = __builtin_amdgcn_wmma_f32_16x16x32_bf16(
            false, a.v, false, b.v, (short)0, acc.v, false, false);
    }

    float* orow = h + (long long)(m0 + (hi << 3)) * FEAT + n0 + r;
#pragma unroll
    for (int rr = 0; rr < 8; ++rr)
        orow[(long long)rr * FEAT] = acc.f[rr];
}

// scalar tail for rows not covered by 16-row tiles (fp32, exact)
__global__ void gemm_tail_kernel(const float* __restrict__ x,
                                 const float* __restrict__ W,
                                 float* __restrict__ h, int row0, int N) {
    int idx = blockIdx.x * blockDim.x + threadIdx.x;
    int nrows = N - row0;
    if (idx >= nrows * FEAT) return;
    int rr = row0 + idx / FEAT, c = idx % FEAT;
    float acc = 0.f;
    for (int k = 0; k < FEAT; ++k) acc += x[(long long)rr * FEAT + k] * W[k * FEAT + c];
    h[(long long)rr * FEAT + c] = acc;
}

// ---------- per-node attention logits ----------
__global__ void __launch_bounds__(256)
att_dots_kernel(const float* __restrict__ h,
                const float* __restrict__ att_src, const float* __restrict__ att_dst,
                float* __restrict__ a_src, float* __restrict__ a_dst) {
    int n = blockIdx.x, t = threadIdx.x;
    int head = t >> 6, c = t & 63;
    float v  = h[(long long)n * FEAT + t];
    __shared__ float s1[256], s2[256];
    s1[t] = v * att_src[head * OUTC + c];
    s2[t] = v * att_dst[head * OUTC + c];
    __syncthreads();
    for (int off = 32; off > 0; off >>= 1) {
        if ((t & 63) < off) { s1[t] += s1[t + off]; s2[t] += s2[t + off]; }
        __syncthreads();
    }
    if ((t & 63) == 0) {
        a_src[n * HEADS + head] = s1[t];
        a_dst[n * HEADS + head] = s2[t];
    }
}

// ---------- segment max over dst (atomic max on ordered keys) ----------
__global__ void edge_max_kernel(const int* __restrict__ ei,
                                const float* __restrict__ a_src,
                                const float* __restrict__ a_dst,
                                unsigned* __restrict__ emax_ord, int E, int Etot) {
    int e = blockIdx.x * blockDim.x + threadIdx.x;
    if (e >= Etot) return;
    int s, d;
    if (e < E) { s = ei[e]; d = ei[E + e]; } else { s = d = e - E; }
#pragma unroll
    for (int hh = 0; hh < HEADS; ++hh) {
        float v = edgeLogit(a_src, a_dst, s, d, hh);
        atomicMax(emax_ord + d * HEADS + hh, ordKey(v));
    }
}

__global__ void emax_convert_kernel(unsigned* __restrict__ emax, int n) {
    int i = blockIdx.x * blockDim.x + threadIdx.x;
    if (i < n) ((float*)emax)[i] = ordToFloat(emax[i]);
}

// ---------- softmax denominators ----------
__global__ void denom_kernel(const int* __restrict__ ei,
                             const float* __restrict__ a_src,
                             const float* __restrict__ a_dst,
                             const float* __restrict__ emaxf,
                             float* __restrict__ denom, int E, int Etot) {
    int e = blockIdx.x * blockDim.x + threadIdx.x;
    if (e >= Etot) return;
    int s, d;
    if (e < E) { s = ei[e]; d = ei[E + e]; } else { s = d = e - E; }
#pragma unroll
    for (int hh = 0; hh < HEADS; ++hh) {
        float v = edgeLogit(a_src, a_dst, s, d, hh);
        atomicAdd(denom + d * HEADS + hh, __expf(v - emaxf[d * HEADS + hh]));
    }
}

// ---------- weighted scatter: one wave per edge, alpha recomputed ----------
__global__ void __launch_bounds__(256)
scatter_kernel(const int* __restrict__ ei, const float* __restrict__ h,
               const float* __restrict__ a_src, const float* __restrict__ a_dst,
               const float* __restrict__ emaxf, const float* __restrict__ denom,
               float* __restrict__ out, int E, int Etot) {
    int wave = threadIdx.x >> 5, lane = threadIdx.x & 31;
    int e = blockIdx.x * 8 + wave;
    if (e >= Etot) return;
    int s, d;
    if (e < E) { s = ei[e]; d = ei[E + e]; } else { s = d = e - E; }

    float alpha[HEADS];
#pragma unroll
    for (int hh = 0; hh < HEADS; ++hh) {
        float v = edgeLogit(a_src, a_dst, s, d, hh);
        alpha[hh] = __expf(v - emaxf[d * HEADS + hh]) / denom[d * HEADS + hh];
    }
    const float* hs = h   + (long long)s * FEAT;
    float*       od = out + (long long)d * FEAT;
#pragma unroll
    for (int j = 0; j < 8; ++j) {
        int c = lane + 32 * j;                 // coalesced 128B per wave
        atomicAdd(od + c, hs[c] * alpha[c >> 6]);
    }
}

// ---------- bias + relu + L2 normalize (in place on d_out) ----------
__global__ void __launch_bounds__(256)
finalize_kernel(float* __restrict__ out, const float* __restrict__ bias) {
    int n = blockIdx.x, t = threadIdx.x;
    float v = fmaxf(out[(long long)n * FEAT + t] + bias[t], 0.f);
    __shared__ float s[256];
    s[t] = v * v;
    __syncthreads();
    for (int off = 128; off > 0; off >>= 1) {
        if (t < off) s[t] += s[t + off];
        __syncthreads();
    }
    float norm = fmaxf(sqrtf(s[0]), 1e-12f);
    out[(long long)n * FEAT + t] = v / norm;
}

// ---------- launch ----------
extern "C" void kernel_launch(void* const* d_in, const int* in_sizes, int n_in,
                              void* d_out, int out_size, void* d_ws, size_t ws_size,
                              hipStream_t stream) {
    const float* x       = (const float*)d_in[0];
    const int*   ei      = (const int*)  d_in[1];
    const float* W       = (const float*)d_in[2];
    const float* att_src = (const float*)d_in[3];
    const float* att_dst = (const float*)d_in[4];
    const float* bias    = (const float*)d_in[5];

    const int N = in_sizes[0] / FEAT;
    const int E = in_sizes[1] / 2;
    const int Etot = E + N;

    // workspace carve-out (~80 MB)
    char* ws = (char*)d_ws;
    size_t cur = 0;
    auto carve = [&](size_t bytes) -> void* {
        void* p = ws + cur;
        cur = (cur + bytes + 255) & ~(size_t)255;
        return p;
    };
    float*          h     = (float*)         carve((size_t)N * FEAT * 4);
    unsigned short* xb    = (unsigned short*)carve((size_t)N * FEAT * 2);
    unsigned short* wt    = (unsigned short*)carve((size_t)FEAT * FEAT * 2);
    float*          asrc  = (float*)         carve((size_t)N * HEADS * 4);
    float*          adst  = (float*)         carve((size_t)N * HEADS * 4);
    unsigned*       emax  = (unsigned*)      carve((size_t)N * HEADS * 4);
    float*          denom = (float*)         carve((size_t)N * HEADS * 4);
    float*          out   = (float*)d_out;
    (void)ws_size; (void)n_in; (void)out_size;

    // init accumulators (0u == 0.0f; emax key 0 < ordKey(-inf) handled by self-loop)
    zero_u32_kernel<<<2048, 256, 0, stream>>>((unsigned*)out, (long long)N * FEAT);
    zero_u32_kernel<<<256, 256, 0, stream>>>(emax, (long long)N * HEADS);
    zero_u32_kernel<<<256, 256, 0, stream>>>((unsigned*)denom, (long long)N * HEADS);

    // bf16 conversions
    cvt_bf16_kernel<<<4096, 256, 0, stream>>>(x, xb, (long long)N * FEAT);
    cvt_wT_kernel<<<(FEAT * FEAT) / 256, 256, 0, stream>>>(W, wt);

    // GEMM (WMMA bf16 -> fp32)
    const int MT = N / 16;
    if (MT > 0)
        gemm_wmma_bf16_kernel<<<MT * 2, 256, 0, stream>>>(xb, wt, h);
    const int row0 = MT * 16;
    if (row0 < N) {
        int tail = (N - row0) * FEAT;
        gemm_tail_kernel<<<(tail + 255) / 256, 256, 0, stream>>>(x, W, h, row0, N);
    }

    // attention logits
    att_dots_kernel<<<N, 256, 0, stream>>>(h, att_src, att_dst, asrc, adst);

    // segment softmax
    edge_max_kernel<<<(Etot + 255) / 256, 256, 0, stream>>>(ei, asrc, adst, emax, E, Etot);
    emax_convert_kernel<<<(N * HEADS + 255) / 256, 256, 0, stream>>>(emax, N * HEADS);
    denom_kernel<<<(Etot + 255) / 256, 256, 0, stream>>>(ei, asrc, adst, (float*)emax, denom, E, Etot);

    // weighted message scatter (one wave per edge)
    scatter_kernel<<<(Etot + 7) / 8, 256, 0, stream>>>(ei, h, asrc, adst, (float*)emax, denom, out, E, Etot);

    // bias + relu + L2 norm
    finalize_kernel<<<N, 256, 0, stream>>>(out, bias);
}
